// StrawberryPCTEncoder_20658792694267
// MI455X (gfx1250) — matrix operand order repaired and checked by reading, not compiled, not run on hardware
//
#include <hip/hip_runtime.h>
#include <math.h>

#define BATCH 8
#define NFULL 8192
#define KNN_K 20

typedef __attribute__((ext_vector_type(16))) __bf16 v16bf;
typedef __attribute__((ext_vector_type(8)))  float  v8f;

struct __align__(16) U4 { unsigned a, b, c, d; };

union Frag16 {
    v16bf    bf;
    unsigned u[8];
    U4       q[2];
};

// ---------------------------------------------------------------------------
// conv1(relu) + conv2 fused: points [B,3,N] -> feat [B,N,128], pts_t [B,N,3]
// ---------------------------------------------------------------------------
__global__ __launch_bounds__(256) void spct_feat(
    const float* __restrict__ points,
    const float* __restrict__ w1, const float* __restrict__ b1,
    const float* __restrict__ w2, const float* __restrict__ b2,
    float* __restrict__ pts_t, float* __restrict__ feat)
{
    __shared__ float w1s[64 * 3], b1s[64], w2s[128 * 64], b2s[128];
    const int tid = threadIdx.x;
    for (int t = tid; t < 192; t += 256) w1s[t] = w1[t];
    if (tid < 64) b1s[tid] = b1[tid];
    for (int t = tid; t < 8192; t += 256) w2s[t] = w2[t];
    if (tid < 128) b2s[tid] = b2[tid];
    __syncthreads();

    const int g = blockIdx.x * 256 + tid;           // 0 .. B*N-1
    const int b = g / NFULL, s = g % NFULL;
    const float* P = points + (long long)b * 3 * NFULL;
    const float p0 = P[s], p1 = P[NFULL + s], p2 = P[2 * NFULL + s];
    float* pt = pts_t + (long long)g * 3;
    pt[0] = p0; pt[1] = p1; pt[2] = p2;

    float x[64];
#pragma unroll
    for (int e = 0; e < 64; ++e) {
        float a = w1s[e * 3] * p0 + w1s[e * 3 + 1] * p1 + w1s[e * 3 + 2] * p2 + b1s[e];
        x[e] = a > 0.f ? a : 0.f;
    }
    float* fo = feat + (long long)g * 128;
    for (int o = 0; o < 128; ++o) {
        float a = b2s[o];
#pragma unroll
        for (int e = 0; e < 64; ++e) a += w2s[o * 64 + e] * x[e];
        fo[o] = a;
    }
}

// ---------------------------------------------------------------------------
// One-shot: convert W2 half of EdgeConv weights (W[o][C:2C]) to bf16 [C][C]
// ---------------------------------------------------------------------------
__global__ void spct_w2bf(const float* __restrict__ W, __bf16* __restrict__ Wbf, int C)
{
    const int g = blockIdx.x * 256 + threadIdx.x;   // C*C
    const int o = g / C, k = g % C;
    Wbf[g] = (__bf16)W[(long long)o * 2 * C + C + k];
}

// ---------------------------------------------------------------------------
// Furthest point sampling: one workgroup per batch, min-dist array in LDS
// ---------------------------------------------------------------------------
template <int NP>
__global__ __launch_bounds__(256) void spct_fps(
    const float* __restrict__ pts, int m, int* __restrict__ idx)
{
    __shared__ float d_s[NP];
    __shared__ float rv[256];  __shared__ int ri[256];
    __shared__ float rv2[32];  __shared__ int ri2[32];
    __shared__ float shp[3];
    const int tid = threadIdx.x, b = blockIdx.x;
    const float* pb = pts + (long long)b * NP * 3;
    const float p0x = pb[0], p0y = pb[1], p0z = pb[2];
    for (int i = tid; i < NP; i += 256) {
        float dx = pb[i * 3] - p0x, dy = pb[i * 3 + 1] - p0y, dz = pb[i * 3 + 2] - p0z;
        d_s[i] = dx * dx + dy * dy + dz * dz;
    }
    if (tid == 0) idx[(long long)b * m] = 0;
    __syncthreads();

    for (int step = 1; step < m; ++step) {
        float bv = -1.f; int bi = 0x7fffffff;
        for (int i = tid; i < NP; i += 256) {
            float v = d_s[i];
            if (v > bv || (v == bv && i < bi)) { bv = v; bi = i; }
        }
        rv[tid] = bv; ri[tid] = bi;
        __syncthreads();
        if (tid < 32) {
            float lv = rv[tid * 8]; int li = ri[tid * 8];
            for (int t = 1; t < 8; ++t) {
                float v = rv[tid * 8 + t]; int ii = ri[tid * 8 + t];
                if (v > lv || (v == lv && ii < li)) { lv = v; li = ii; }
            }
            rv2[tid] = lv; ri2[tid] = li;
        }
        __syncthreads();
        if (tid == 0) {
            float lv = rv2[0]; int li = ri2[0];
            for (int t = 1; t < 32; ++t) {
                float v = rv2[t]; int ii = ri2[t];
                if (v > lv || (v == lv && ii < li)) { lv = v; li = ii; }
            }
            idx[(long long)b * m + step] = li;
            shp[0] = pb[li * 3]; shp[1] = pb[li * 3 + 1]; shp[2] = pb[li * 3 + 2];
        }
        __syncthreads();
        const float nx = shp[0], ny = shp[1], nz = shp[2];
        for (int i = tid; i < NP; i += 256) {
            float dx = pb[i * 3] - nx, dy = pb[i * 3 + 1] - ny, dz = pb[i * 3 + 2] - nz;
            float dd = dx * dx + dy * dy + dz * dz;
            if (dd < d_s[i]) d_s[i] = dd;
        }
        __syncthreads();
    }
}

// ---------------------------------------------------------------------------
// kNN (K=20), register top-k with unrolled compare-swap chain
// ---------------------------------------------------------------------------
template <int NP>
__global__ __launch_bounds__(256) void spct_knn(
    const float* __restrict__ pts, int* __restrict__ nbr)
{
    __shared__ float cx[256], cy[256], cz[256];
    const int tid = threadIdx.x;
    const int bpb = NP / 256;
    const int b = blockIdx.x / bpb;
    const int q = (blockIdx.x % bpb) * 256 + tid;
    const float* pb = pts + (long long)b * NP * 3;
    const float qx = pb[q * 3], qy = pb[q * 3 + 1], qz = pb[q * 3 + 2];

    float bd[KNN_K]; int bi[KNN_K];
#pragma unroll
    for (int t = 0; t < KNN_K; ++t) { bd[t] = 3.0e38f; bi[t] = 0; }

    for (int t0 = 0; t0 < NP; t0 += 256) {
        __syncthreads();
        const int cI = t0 + tid;
        cx[tid] = pb[cI * 3]; cy[tid] = pb[cI * 3 + 1]; cz[tid] = pb[cI * 3 + 2];
        __syncthreads();
        for (int jj = 0; jj < 256; ++jj) {
            const int j = t0 + jj;
            float dx = cx[jj] - qx, dy = cy[jj] - qy, dz = cz[jj] - qz;
            float d = dx * dx + dy * dy + dz * dz;
            if (j != q && d < bd[KNN_K - 1]) {
                bd[KNN_K - 1] = d; bi[KNN_K - 1] = j;
#pragma unroll
                for (int p = KNN_K - 1; p > 0; --p) {
                    if (bd[p] < bd[p - 1]) {
                        float td = bd[p]; bd[p] = bd[p - 1]; bd[p - 1] = td;
                        int   ti = bi[p]; bi[p] = bi[p - 1]; bi[p - 1] = ti;
                    }
                }
            }
        }
    }
    const long long base = ((long long)b * NP + q) * KNN_K;
#pragma unroll
    for (int t = 0; t < KNN_K; ++t) nbr[base + t] = bi[t];
}

// ---------------------------------------------------------------------------
// Gather rows: dst[b,i,:] = src[b, idx[b,i], :]
// ---------------------------------------------------------------------------
__global__ void spct_gather(const float* __restrict__ src, const int* __restrict__ idx,
                            float* __restrict__ dst, int n_in, int n_out, int Cc)
{
    const int r = blockIdx.x;
    const int b = r / n_out, i = r % n_out;
    const int s = idx[(long long)b * n_out + i];
    const float* sp = src + ((long long)b * n_in + s) * Cc;
    float* dp = dst + (long long)r * Cc;
    for (int c = threadIdx.x; c < Cc; c += blockDim.x) dp[c] = sp[c];
}

// ---------------------------------------------------------------------------
// EdgeConv via WMMA bf16: out = [xi , max_j relu(W1*xi + W2*(xj-xi) + b)]
// W1*xi hoisted (rank-1 over neighbors); WMMA covers the (xj-xi) half with
// pre-converted bf16 weights (fragment = 16 contiguous bf16 per lane).
// One workgroup (4 waves) per point; 20 neighbors padded to 32 rows.
// ---------------------------------------------------------------------------
template <int C, int NP>
__global__ __launch_bounds__(128) void spct_edge(
    const float* __restrict__ f, const int* __restrict__ nbr,
    const float* __restrict__ W, const __bf16* __restrict__ Wbf,
    const float* __restrict__ bias, float* __restrict__ xout)
{
    __shared__ __align__(16) __bf16 As[32][C + 8];   // +8 halves => conflict-free b128
    __shared__ float Red[4][32][16];
    __shared__ float xi_s[C];
    __shared__ float base_s[C];
    __shared__ int   nbs[32];

    const int tid  = threadIdx.x;
    const int wave = tid >> 5, lane = tid & 31;
    const int gi = blockIdx.x;
    const int b = gi / NP, i = gi % NP;
    const long long rowi = (long long)b * NP + i;
    const float* fi = f + rowi * C;
    float* orow = xout + rowi * (2 * C);

    if (tid < 32) nbs[tid] = (tid < KNN_K) ? nbr[rowi * KNN_K + tid] : 0;
    for (int c = tid; c < C; c += 128) { float v = fi[c]; xi_s[c] = v; orow[c] = v; }
    __syncthreads();

    // base_s[o] = W[o][0:C] . xi + bias[o]
    for (int o = tid; o < C; o += 128) {
        const float* wrow = W + (long long)o * (2 * C);
        float s = 0.f;
        for (int c = 0; c < C; ++c) s += wrow[c] * xi_s[c];
        base_s[o] = s + bias[o];
    }
    // A tile rows = neighbors (bf16), cols = (xj - xi); native bf16 cvt
    for (int t = tid; t < 32 * C; t += 128) {
        const int j = t / C, c = t % C;
        float v = 0.f;
        if (j < KNN_K) v = f[((long long)b * NP + nbs[j]) * C + c] - xi_s[c];
        As[j][c] = (__bf16)v;
    }
    __syncthreads();

    const int ntiles = C / 16;                 // 8 / 16 / 32 -> uniform over 4 waves
    const int half = lane >> 4;                // 0/1
    const int nidx = lane & 15;
    for (int nt = wave; nt < ntiles; nt += 4) {
        v8f acc0 = {}; v8f acc1 = {};
        const __bf16* wrow = Wbf + (long long)(nt * 16 + nidx) * C;  // bf16 W2 row
        for (int ks = 0; ks < C; ks += 32) {
            const U4* wq = (const U4*)(wrow + ks + half * 16);       // 16 contiguous bf16
            __builtin_prefetch((const void*)(wq + 4), 0, 1);         // global_prefetch_b8
            Frag16 bfrag;
            bfrag.q[0] = wq[0];
            bfrag.q[1] = wq[1];
            const int co = ks + half * 16;                           // A fragment: 16 contiguous
            Frag16 a0, a1;
            a0.q[0] = *(const U4*)&As[nidx][co];
            a0.q[1] = *(const U4*)&As[nidx][co + 8];
            a1.q[0] = *(const U4*)&As[16 + nidx][co];
            a1.q[1] = *(const U4*)&As[16 + nidx][co + 8];
            acc0 = __builtin_amdgcn_wmma_f32_16x16x32_bf16(false, a0.bf, false, bfrag.bf,
                                                           (short)0, acc0, false, false);
            acc1 = __builtin_amdgcn_wmma_f32_16x16x32_bf16(false, a1.bf, false, bfrag.bf,
                                                           (short)0, acc1, false, false);
        }
#pragma unroll
        for (int r = 0; r < 8; ++r) {
            Red[wave][r + 8 * half][nidx]      = acc0[r];
            Red[wave][16 + r + 8 * half][nidx] = acc1[r];
        }
        __syncthreads();
        if (lane < 16) {
            const int o2 = nt * 16 + lane;
            const float bb = base_s[o2];
            float m = 0.f;                       // relu => max >= 0
#pragma unroll
            for (int j = 0; j < KNN_K; ++j) {
                float v = bb + Red[wave][j][lane];
                m = v > m ? v : m;
            }
            orow[C + o2] = m;
        }
        __syncthreads();
    }
}

// ---------------------------------------------------------------------------
// Global max pool: x3 [B,512,1024] -> out [B,1024]
// ---------------------------------------------------------------------------
__global__ void spct_maxpool(const float* __restrict__ x3, float* __restrict__ out)
{
    const int g = blockIdx.x * 256 + threadIdx.x;   // B*1024
    const int b = g / 1024, c = g % 1024;
    const float* xp = x3 + (long long)b * 512 * 1024 + c;
    float m = -3.0e38f;
    for (int r = 0; r < 512; ++r) { float v = xp[r * 1024]; m = v > m ? v : m; }
    out[g] = m;
}

// ---------------------------------------------------------------------------
// Tail: MLP chain + 3 single-token transformer blocks + output heads
// ---------------------------------------------------------------------------
struct SAParams {
    const float *proj_w, *proj_b, *in_w, *in_b, *out_w, *out_b;
    const float *w11, *b11, *w12, *b12, *ln12_g, *ln12_b, *ln13_g, *ln13_b;
};
struct TailArgs {
    const float *ps_adj_w, *ps_adj_b, *ps_w, *ps_b, *ps_refuse_w, *ps_refuse_b;
    SAParams sa[3];
    const float *out1_w, *out1_b, *out_w, *out_b;
    const float *xg;
    float *fine;
};

__global__ __launch_bounds__(256) void spct_tail(TailArgs A)
{
    __shared__ float sbuf[1024];
    __shared__ float hbuf[256];
    __shared__ float tok[64], t1[64], t2[64], t3[64];
    __shared__ float mv[2];
    __shared__ float f3v[3];
    const int tid = threadIdx.x, b = blockIdx.x;
    const float* xg = A.xg + (long long)b * 1024;

    { // h1 = relu(ps_adj_w @ xg + b)   [256]
        const float* w = A.ps_adj_w + (long long)tid * 1024;
        float s = 0.f;
        for (int i = 0; i < 1024; ++i) s += w[i] * xg[i];
        s += A.ps_adj_b[tid];
        hbuf[tid] = s > 0.f ? s : 0.f;
    }
    __syncthreads();
    if (tid < 128) { // h2 [128]
        const float* w = A.ps_w + (long long)tid * 256;
        float s = 0.f;
        for (int i = 0; i < 256; ++i) s += w[i] * hbuf[i];
        s += A.ps_b[tid];
        sbuf[tid] = s > 0.f ? s : 0.f;
    }
    __syncthreads();
    if (tid < 64) { // h3 [64] -> token
        const float* w = A.ps_refuse_w + (long long)tid * 128;
        float s = 0.f;
        for (int i = 0; i < 128; ++i) s += w[i] * sbuf[i];
        s += A.ps_refuse_b[tid];
        tok[tid] = s > 0.f ? s : 0.f;
    }
    __syncthreads();

    for (int blk = 0; blk < 3; ++blk) {
        const SAParams p = A.sa[blk];
        if (tid < 64) { // proj
            const float* w = p.proj_w + tid * 64;
            float s = 0.f;
            for (int i = 0; i < 64; ++i) s += w[i] * tok[i];
            t1[tid] = s + p.proj_b[tid];
        }
        __syncthreads();
        if (tid == 0) {
            float m = 0.f; for (int i = 0; i < 64; ++i) m += t1[i];
            m *= (1.f / 64.f);
            float v = 0.f; for (int i = 0; i < 64; ++i) { float d = t1[i] - m; v += d * d; }
            mv[0] = m; mv[1] = v * (1.f / 64.f);
        }
        __syncthreads();
        if (tid < 64)   // ln13 -> s1
            t2[tid] = (t1[tid] - mv[0]) * rsqrtf(mv[1] + 1e-5f) * p.ln13_g[tid] + p.ln13_b[tid];
        __syncthreads();
        if (tid < 64) { // S=1 attention == V projection
            const float* w = p.in_w + (128 + tid) * 64;
            float s = 0.f;
            for (int i = 0; i < 64; ++i) s += w[i] * t2[i];
            t1[tid] = s + p.in_b[128 + tid];
        }
        __syncthreads();
        if (tid < 64) { // out proj + residual
            const float* w = p.out_w + tid * 64;
            float s = 0.f;
            for (int i = 0; i < 64; ++i) s += w[i] * t1[i];
            t3[tid] = t2[tid] + s + p.out_b[tid];
        }
        __syncthreads();
        if (tid == 0) {
            float m = 0.f; for (int i = 0; i < 64; ++i) m += t3[i];
            m *= (1.f / 64.f);
            float v = 0.f; for (int i = 0; i < 64; ++i) { float d = t3[i] - m; v += d * d; }
            mv[0] = m; mv[1] = v * (1.f / 64.f);
        }
        __syncthreads();
        if (tid < 64)   // ln12
            tok[tid] = (t3[tid] - mv[0]) * rsqrtf(mv[1] + 1e-5f) * p.ln12_g[tid] + p.ln12_b[tid];
        __syncthreads();
        for (int ff = tid; ff < 1024; ff += 256) { // FFN up + exact GELU
            const float* w = p.w11 + (long long)ff * 64;
            float s = 0.f;
            for (int i = 0; i < 64; ++i) s += w[i] * tok[i];
            s += p.b11[ff];
            sbuf[ff] = 0.5f * s * (1.f + erff(s * 0.70710678118654752f));
        }
        __syncthreads();
        if (tid < 64) { // FFN down + residual
            const float* w = p.w12 + (long long)tid * 1024;
            float s = 0.f;
            for (int i = 0; i < 1024; ++i) s += w[i] * sbuf[i];
            t1[tid] = s + p.b12[tid] + tok[tid];
        }
        __syncthreads();
        if (tid < 64) tok[tid] = t1[tid];
        __syncthreads();
    }

    { // out1 (relu) [256]
        const float* w = A.out1_w + (long long)tid * 64;
        float s = 0.f;
        for (int i = 0; i < 64; ++i) s += w[i] * tok[i];
        s += A.out1_b[tid];
        sbuf[tid] = s > 0.f ? s : 0.f;
    }
    __syncthreads();
    if (tid < 3) { // out [3]
        const float* w = A.out_w + tid * 256;
        float s = 0.f;
        for (int i = 0; i < 256; ++i) s += w[i] * sbuf[i];
        f3v[tid] = s + A.out_b[tid];
    }
    __syncthreads();
    for (int t = tid; t < 3 * 128; t += 256) { // broadcast token -> 128 columns
        const int r = t / 128, m2 = t % 128;
        A.fine[((long long)b * 3 + r) * 128 + m2] = f3v[r];
    }
}

// ---------------------------------------------------------------------------
extern "C" void kernel_launch(void* const* d_in, const int* in_sizes, int n_in,
                              void* d_out, int out_size, void* d_ws, size_t ws_size,
                              hipStream_t stream)
{
    (void)in_sizes; (void)n_in; (void)out_size; (void)ws_size;
    auto F = [&](int i) { return (const float*)d_in[i]; };

    size_t off = 0;
    auto alloc = [&](size_t elems) -> void* {       // elems in float units
        void* p = (char*)d_ws + off;
        off += elems * sizeof(float);
        return p;
    };
    float*  pts_t = (float*)alloc((size_t)8 * 8192 * 3);
    float*  feat  = (float*)alloc((size_t)8 * 8192 * 128);
    int*    idx0  = (int*)  alloc((size_t)8 * 2048);
    float*  pts0  = (float*)alloc((size_t)8 * 2048 * 3);
    float*  f0    = (float*)alloc((size_t)8 * 2048 * 128);
    int*    nbr0  = (int*)  alloc((size_t)8 * 2048 * 20);
    float*  x1    = (float*)alloc((size_t)8 * 2048 * 256);
    int*    idx1  = (int*)  alloc((size_t)8 * 1024);
    float*  pts1  = (float*)alloc((size_t)8 * 1024 * 3);
    float*  f1    = (float*)alloc((size_t)8 * 1024 * 256);
    int*    nbr1  = (int*)  alloc((size_t)8 * 1024 * 20);
    float*  x2    = (float*)alloc((size_t)8 * 1024 * 512);
    int*    idx2  = (int*)  alloc((size_t)8 * 512);
    float*  pts2  = (float*)alloc((size_t)8 * 512 * 3);
    float*  f2    = (float*)alloc((size_t)8 * 512 * 512);
    int*    nbr2  = (int*)  alloc((size_t)8 * 512 * 20);
    float*  x3    = (float*)alloc((size_t)8 * 512 * 1024);
    __bf16* wbf1  = (__bf16*)alloc((size_t)128 * 128 / 2);   // bf16: half a float each
    __bf16* wbf2  = (__bf16*)alloc((size_t)256 * 256 / 2);
    __bf16* wbf3  = (__bf16*)alloc((size_t)512 * 512 / 2);

    spct_feat<<<256, 256, 0, stream>>>(F(0), F(1), F(2), F(3), F(4), pts_t, feat);

    // one-shot bf16 weight conversion for the three EdgeConv stages
    spct_w2bf<<<128 * 128 / 256, 256, 0, stream>>>(F(5), wbf1, 128);
    spct_w2bf<<<256 * 256 / 256, 256, 0, stream>>>(F(7), wbf2, 256);
    spct_w2bf<<<512 * 512 / 256, 256, 0, stream>>>(F(9), wbf3, 512);

    // stage 1: N -> 2048, C 128 -> 256
    spct_fps<8192><<<8, 256, 0, stream>>>(pts_t, 2048, idx0);
    spct_gather<<<8 * 2048, 32, 0, stream>>>(pts_t, idx0, pts0, 8192, 2048, 3);
    spct_gather<<<8 * 2048, 128, 0, stream>>>(feat, idx0, f0, 8192, 2048, 128);
    spct_knn<2048><<<8 * 8, 256, 0, stream>>>(pts0, nbr0);
    spct_edge<128, 2048><<<8 * 2048, 128, 0, stream>>>(f0, nbr0, F(5), wbf1, F(6), x1);

    // stage 2: 2048 -> 1024, C 256 -> 512
    spct_fps<2048><<<8, 256, 0, stream>>>(pts0, 1024, idx1);
    spct_gather<<<8 * 1024, 32, 0, stream>>>(pts0, idx1, pts1, 2048, 1024, 3);
    spct_gather<<<8 * 1024, 128, 0, stream>>>(x1, idx1, f1, 2048, 1024, 256);
    spct_knn<1024><<<8 * 4, 256, 0, stream>>>(pts1, nbr1);
    spct_edge<256, 1024><<<8 * 1024, 128, 0, stream>>>(f1, nbr1, F(7), wbf2, F(8), x2);

    // stage 3: 1024 -> 512, C 512 -> 1024
    spct_fps<1024><<<8, 256, 0, stream>>>(pts1, 512, idx2);
    spct_gather<<<8 * 512, 32, 0, stream>>>(pts1, idx2, pts2, 1024, 512, 3);
    spct_gather<<<8 * 512, 128, 0, stream>>>(x2, idx2, f2, 1024, 512, 512);
    spct_knn<512><<<8 * 2, 256, 0, stream>>>(pts2, nbr2);
    spct_edge<512, 512><<<8 * 512, 128, 0, stream>>>(f2, nbr2, F(9), wbf3, F(10), x3);

    // global max pool -> first output chunk (x_g, 8192 floats)
    spct_maxpool<<<32, 256, 0, stream>>>(x3, (float*)d_out);

    // tail (reads x_g from d_out, writes fine after it)
    TailArgs A;
    A.ps_adj_w = F(11); A.ps_adj_b = F(12);
    A.ps_w = F(13); A.ps_b = F(14);
    A.ps_refuse_w = F(15); A.ps_refuse_b = F(16);
    for (int k = 0; k < 3; ++k) {
        const int base = 17 + k * 14;
        A.sa[k].proj_w = F(base + 0);  A.sa[k].proj_b = F(base + 1);
        A.sa[k].in_w   = F(base + 2);  A.sa[k].in_b   = F(base + 3);
        A.sa[k].out_w  = F(base + 4);  A.sa[k].out_b  = F(base + 5);
        A.sa[k].w11    = F(base + 6);  A.sa[k].b11    = F(base + 7);
        A.sa[k].w12    = F(base + 8);  A.sa[k].b12    = F(base + 9);
        A.sa[k].ln12_g = F(base + 10); A.sa[k].ln12_b = F(base + 11);
        A.sa[k].ln13_g = F(base + 12); A.sa[k].ln13_b = F(base + 13);
    }
    A.out1_w = F(59); A.out1_b = F(60);
    A.out_w  = F(61); A.out_b  = F(62);
    A.xg   = (const float*)d_out;
    A.fine = (float*)d_out + 8192;
    spct_tail<<<8, 256, 0, stream>>>(A);
}
